// CMoE_25271587570017
// MI455X (gfx1250) — matrix operand order repaired and checked by reading, not compiled, not run on hardware
//
#include <hip/hip_runtime.h>

// ---------------------------------------------------------------------------
// CDNA5 (gfx1250) MoE pipeline: every conv/FC is a bf16 WMMA implicit GEMM
// using v_wmma_f32_16x16x32_bf16 (wave32).
//  - Activations NHWC bf16 with ZERO-PADDED halos -> branch-free A loads
//    (padding handled by memory, not EXEC masking).
//  - Deconv stride-2 parity handled by v_cndmask on the address (invalid
//    lanes read the zero halo), keeping EXEC all-ones for WMMA.
//  - Each wave computes TWO 16x16 N-tiles per A fragment (2 accumulators)
//    to double wmma density per byte loaded.
//  - Top-2 routing: expert blocks early-out (block-uniform) for unrouted
//    batches; tiles never cross a batch (144 px = 9 tiles exactly).
// ---------------------------------------------------------------------------

typedef __attribute__((ext_vector_type(16))) __bf16 v16bf;
typedef __attribute__((ext_vector_type(8)))  __bf16 v8bf;
typedef __attribute__((ext_vector_type(8)))  float  v8f;

#define DEV __device__ __forceinline__

// ---- bf16 helpers ----------------------------------------------------------
DEV __bf16 f2bf(float f) {
  union { float f; unsigned u; } v; v.f = f;
  unsigned r = v.u + 0x7fffu + ((v.u >> 16) & 1u);   // round-to-nearest-even
  union { unsigned short s; __bf16 b; } o; o.s = (unsigned short)(r >> 16);
  return o.b;
}
DEV float bf2f(__bf16 b) {
  union { unsigned short s; __bf16 b; } i; i.b = b;
  union { float f; unsigned u; } o; o.u = ((unsigned)i.s) << 16;
  return o.f;
}

// ---- WMMA fragment loaders (branch-free) -----------------------------------
// A (16x32, M x K): lane<16 -> row=lane,   K elements {0..15}
//                   lane>=16 -> row=lane-16, K elements {16..31}
DEV v16bf load_a_frag(const __bf16* rowBase, int half) {
  v8bf lo = *(const v8bf*)(rowBase + half * 16);         // 16B aligned
  v8bf hi = *(const v8bf*)(rowBase + half * 16 + 8);
  v16bf a;
#pragma unroll
  for (int i = 0; i < 8; ++i) { a[i] = lo[i]; a[i + 8] = hi[i]; }
  return a;
}

// B (32x16, K x N) pre-packed: element (lane, i) = W[k=(lane>>4)*16+i, n=lane&15]
DEV v16bf load_b_frag(const __bf16* tile, int lane) {
  return *(const v16bf*)(tile + lane * 16);              // one 32B load
}

DEV v8f wmma_bf16(v16bf a, v16bf b, v8f c) {
  return __builtin_amdgcn_wmma_f32_16x16x32_bf16(false, a, false, b,
                                                 (short)0, c, false, false);
}

// ---------------------------------------------------------------------------
static constexpr int B_   = 1024;
static constexpr int CIN  = 128;
static constexpr int COUT = 64;
static constexpr int E_   = 8;
static constexpr int SPIX = 36;     // 6x6 (unpadded)
static constexpr int XPAD = 64;     // 8x8 padded pixels
static constexpr int OPIX = 144;    // 12x12 (unpadded)
static constexpr int YPAD = 196;    // 14x14 padded pixels

// ---------------------------------------------------------------------------
__global__ void k_zero_f32(float* p, int n) {
  int stride = gridDim.x * blockDim.x;
  for (int i = blockIdx.x * blockDim.x + threadIdx.x; i < n; i += stride)
    p[i] = 0.0f;
}
__global__ void k_zero_u32(unsigned* p, int n) {
  int stride = gridDim.x * blockDim.x;
  for (int i = blockIdx.x * blockDim.x + threadIdx.x; i < n; i += stride)
    p[i] = 0u;
}

// ---------------------------------------------------------------------------
// pack x: NCHW f32 -> padded NHWC bf16  xp[b][ph:8][pw:8][ci], zero halo
// ---------------------------------------------------------------------------
__global__ void k_pack_x(const float* x, __bf16* xp) {
  int t = blockIdx.x * blockDim.x + threadIdx.x;
  if (t >= B_ * XPAD * CIN) return;
  int ci = t & 127;
  int pp = (t >> 7) & 63;
  int b  = t >> 13;
  int ih = (pp >> 3) - 1, iw = (pp & 7) - 1;
  float v = 0.0f;
  if ((unsigned)ih < 6u && (unsigned)iw < 6u)
    v = x[(b * CIN + ci) * SPIX + ih * 6 + iw];
  xp[t] = f2bf(v);
}

// ---------------------------------------------------------------------------
// Weight packers (fragment layout)
// ---------------------------------------------------------------------------
__global__ void k_pack_gw(const float* gw, __bf16* Bg) {   // [tap][kb:4][nb:8][512]
  int t = blockIdx.x * blockDim.x + threadIdx.x;
  if (t >= 288 * 512) return;
  int i = t & 15, lane = (t >> 4) & 31, tile = t >> 9;
  int nb = tile & 7, kb = (tile >> 3) & 3, tap = tile >> 5;
  int kh = tap / 3, kw = tap % 3;
  int ci = kb * 32 + ((lane >> 4) << 4) + i;
  int co = nb * 16 + (lane & 15);
  Bg[t] = f2bf(gw[((co * CIN + ci) * 3 + kh) * 3 + kw]);
}

__global__ void k_pack_fc1(const float* fw, __bf16* Bf) {  // [kb:36][nb:16][512]
  int t = blockIdx.x * blockDim.x + threadIdx.x;
  if (t >= 576 * 512) return;
  int i = t & 15, lane = (t >> 4) & 31, tile = t >> 9;
  int nb = tile & 15, kb = tile >> 4;
  int k = kb * 32 + ((lane >> 4) << 4) + i;
  int n = nb * 16 + (lane & 15);
  Bf[t] = f2bf(fw[n * 1152 + k]);
}

__global__ void k_pack_wd(const float* wd, __bf16* Bd) {   // [e][tap][kb:4][nb:4][512]
  int t = blockIdx.x * blockDim.x + threadIdx.x;
  if (t >= 1152 * 512) return;
  int i = t & 15, lane = (t >> 4) & 31, tile = t >> 9;
  int nb = tile & 3, kb = (tile >> 2) & 3, tap = (tile >> 4) % 9, e = tile / 144;
  int kh = tap / 3, kw = tap % 3;
  int ci = kb * 32 + ((lane >> 4) << 4) + i;
  int co = nb * 16 + (lane & 15);
  Bd[t] = f2bf(wd[(((e * CIN + ci) * COUT + co) * 3 + kh) * 3 + kw]);
}

__global__ void k_pack_wc(const float* wc, __bf16* Bc) {   // [e][tap][kb:2][nb:4][512]
  int t = blockIdx.x * blockDim.x + threadIdx.x;
  if (t >= 576 * 512) return;
  int i = t & 15, lane = (t >> 4) & 31, tile = t >> 9;
  int nb = tile & 3, kb = (tile >> 2) & 1, tap = (tile >> 3) % 9, e = tile / 72;
  int kh = tap / 3, kw = tap % 3;
  int ci = kb * 32 + ((lane >> 4) << 4) + i;
  int co = nb * 16 + (lane & 15);
  Bc[t] = f2bf(wc[(((e * COUT + co) * COUT + ci) * 3 + kh) * 3 + kw]);
}

// ---------------------------------------------------------------------------
// Gate conv 3x3 pad1 (128->128) + ReLU.  Branch-free (padded input).
// grid 2304, block 128 (4 waves); wave w -> n-tiles {2w, 2w+1} (N=128).
// ---------------------------------------------------------------------------
__global__ void k_gate_conv(const __bf16* __restrict__ xp,
                            const __bf16* __restrict__ Bg,
                            const float* __restrict__ gb,
                            __bf16* __restrict__ hp) {
  const int lane = threadIdx.x & 31;
  const int wv   = threadIdx.x >> 5;
  const int nb0  = wv * 2, nb1 = nb0 + 1;
  const int mt   = blockIdx.x;
  const int half = lane >> 4;
  const int gA   = mt * 16 + (lane & 15);
  const int b    = gA / SPIX, pix = gA % SPIX;
  const int h    = pix / 6,   w   = pix % 6;

  v8f acc0 = {}, acc1 = {};
#pragma unroll
  for (int kh = 0; kh < 3; ++kh) {
#pragma unroll
    for (int kw = 0; kw < 3; ++kw) {
      // padded coords: (h+kh-1)+1 = h+kh  in [0,7]
      const __bf16* rowBase = xp + ((b * XPAD + (h + kh) * 8 + (w + kw)) << 7);
      const int tap = kh * 3 + kw;
#pragma unroll
      for (int kb = 0; kb < 4; ++kb) {
        v16bf a  = load_a_frag(rowBase + kb * 32, half);
        v16bf b0 = load_b_frag(Bg + ((tap * 4 + kb) * 8 + nb0) * 512, lane);
        v16bf b1 = load_b_frag(Bg + ((tap * 4 + kb) * 8 + nb1) * 512, lane);
        acc0 = wmma_bf16(a, b0, acc0);
        acc1 = wmma_bf16(a, b1, acc1);
      }
    }
  }
  const int co0 = nb0 * 16 + (lane & 15);
  const int co1 = nb1 * 16 + (lane & 15);
  const float bias0 = gb[co0], bias1 = gb[co1];
#pragma unroll
  for (int r = 0; r < 8; ++r) {
    int gC = mt * 16 + r + (half << 3);
    float v0 = acc0[r] + bias0; v0 = v0 > 0.0f ? v0 : 0.0f;
    float v1 = acc1[r] + bias1; v1 = v1 > 0.0f ? v1 : 0.0f;
    hp[(gC << 7) + co0] = f2bf(v0);
    hp[(gC << 7) + co1] = f2bf(v1);
  }
}

// ---------------------------------------------------------------------------
// 2x2 maxpool + torch flatten  pfl[b][ci*9 + ph*3 + pw]
// ---------------------------------------------------------------------------
__global__ void k_maxpool(const __bf16* __restrict__ hp, __bf16* __restrict__ pfl) {
  int t = blockIdx.x * blockDim.x + threadIdx.x;
  if (t >= B_ * 1152) return;
  int pw = t % 3, ph = (t / 3) % 3, ci = (t / 9) % 128, b = t / 1152;
  float m = -3.0e38f;
#pragma unroll
  for (int dh = 0; dh < 2; ++dh)
#pragma unroll
    for (int dw = 0; dw < 2; ++dw) {
      float v = bf2f(hp[((b * SPIX + (2 * ph + dh) * 6 + (2 * pw + dw)) << 7) + ci]);
      m = v > m ? v : m;
    }
  pfl[t] = f2bf(m);
}

// ---------------------------------------------------------------------------
// fc1 [1024,1152] x W^T + ReLU.  grid (64,2), block 128; wave -> 2 n-tiles.
// ---------------------------------------------------------------------------
__global__ void k_fc1(const __bf16* __restrict__ pfl,
                      const __bf16* __restrict__ Bf,
                      const float* __restrict__ fb,
                      __bf16* __restrict__ zp) {
  const int lane = threadIdx.x & 31;
  const int nb0  = blockIdx.y * 8 + (threadIdx.x >> 5) * 2;
  const int nb1  = nb0 + 1;
  const int mt   = blockIdx.x;
  const int half = lane >> 4;
  const int row  = mt * 16 + (lane & 15);

  v8f acc0 = {}, acc1 = {};
  const __bf16* aBase = pfl + row * 1152;
#pragma unroll 4
  for (int kb = 0; kb < 36; ++kb) {
    v16bf a  = load_a_frag(aBase + kb * 32, half);
    v16bf b0 = load_b_frag(Bf + (kb * 16 + nb0) * 512, lane);
    v16bf b1 = load_b_frag(Bf + (kb * 16 + nb1) * 512, lane);
    acc0 = wmma_bf16(a, b0, acc0);
    acc1 = wmma_bf16(a, b1, acc1);
  }
  const int n0 = nb0 * 16 + (lane & 15);
  const int n1 = nb1 * 16 + (lane & 15);
  const float bias0 = fb[n0], bias1 = fb[n1];
#pragma unroll
  for (int r = 0; r < 8; ++r) {
    int rowC = mt * 16 + r + (half << 3);
    float v0 = acc0[r] + bias0; v0 = v0 > 0.0f ? v0 : 0.0f;
    float v1 = acc1[r] + bias1; v1 = v1 > 0.0f ? v1 : 0.0f;
    zp[rowC * 256 + n0] = f2bf(v0);
    zp[rowC * 256 + n1] = f2bf(v1);
  }
}

// ---------------------------------------------------------------------------
// fc2 (256->8) + top-2 + softmax -> dense routing weights wgt[b][8]
// ---------------------------------------------------------------------------
__global__ void k_route(const __bf16* __restrict__ zp,
                        const float* __restrict__ fw,
                        const float* __restrict__ fb,
                        float* __restrict__ wgt) {
  int b = blockIdx.x * blockDim.x + threadIdx.x;
  if (b >= B_) return;
  float l[8];
#pragma unroll
  for (int e = 0; e < 8; ++e) {
    float s = fb[e];
    for (int k = 0; k < 256; ++k)
      s += bf2f(zp[b * 256 + k]) * fw[e * 256 + k];
    l[e] = s;
  }
  int i1 = 0; float m1 = l[0];
#pragma unroll
  for (int e = 1; e < 8; ++e) if (l[e] > m1) { m1 = l[e]; i1 = e; }
  int i2 = -1; float m2 = -3.0e38f;
#pragma unroll
  for (int e = 0; e < 8; ++e) if (e != i1 && l[e] > m2) { m2 = l[e]; i2 = e; }
  float e2 = __expf(m2 - m1);
  float inv = 1.0f / (1.0f + e2);
#pragma unroll
  for (int e = 0; e < 8; ++e) wgt[b * 8 + e] = 0.0f;
  wgt[b * 8 + i1] = inv;
  wgt[b * 8 + i2] = e2 * inv;
}

// ---------------------------------------------------------------------------
// Expert transposed conv (128->64, 6x6 -> 12x12, k3 s2 p1 op1) + ReLU.
// grid 9216, block 64 (2 waves); wave -> 2 n-tiles (N=64).
// Stride-2 parity handled by address select into the zero halo (no EXEC div).
// Output written into padded y1p[b][14][14][64].
// ---------------------------------------------------------------------------
__global__ void k_deconv(const __bf16* __restrict__ xp,
                         const __bf16* __restrict__ Bd,
                         const float* __restrict__ bd,
                         const float* __restrict__ wgt,
                         __bf16* __restrict__ y1p, int e) {
  const int b  = blockIdx.x / 9;
  const int mt = blockIdx.x % 9;
  if (wgt[b * 8 + e] == 0.0f) return;     // block-uniform early out

  const int lane = threadIdx.x & 31;
  const int wv   = threadIdx.x >> 5;
  const int nb0  = wv * 2, nb1 = nb0 + 1;
  const int half = lane >> 4;
  const int pix  = mt * 16 + (lane & 15);
  const int oh   = pix / 12, ow = pix % 12;

  v8f acc0 = {}, acc1 = {};
#pragma unroll
  for (int kh = 0; kh < 3; ++kh) {
    int nh = oh + 1 - kh;                 // 2*ih = nh ; valid iff nh even
    int ih = nh >> 1;                     // in [-?..6]; padded index ih+1 in [0,7]
#pragma unroll
    for (int kw = 0; kw < 3; ++kw) {
      int nw = ow + 1 - kw;
      int iw = nw >> 1;
      bool valid = (((nh | nw) & 1) == 0);
      // invalid lanes -> batch's halo corner (pixel 0 is zero padding)
      int off = valid ? ((b * XPAD + (ih + 1) * 8 + (iw + 1)) << 7) : (b << 13);
      const __bf16* rowBase = xp + off;
      const int tap = kh * 3 + kw;
#pragma unroll
      for (int kb = 0; kb < 4; ++kb) {
        v16bf a  = load_a_frag(rowBase + kb * 32, half);
        v16bf b0 = load_b_frag(Bd + (((e * 9 + tap) * 4 + kb) * 4 + nb0) * 512, lane);
        v16bf b1 = load_b_frag(Bd + (((e * 9 + tap) * 4 + kb) * 4 + nb1) * 512, lane);
        acc0 = wmma_bf16(a, b0, acc0);
        acc1 = wmma_bf16(a, b1, acc1);
      }
    }
  }
  const int co0 = nb0 * 16 + (lane & 15);
  const int co1 = nb1 * 16 + (lane & 15);
  const float bias0 = bd[e * COUT + co0], bias1 = bd[e * COUT + co1];
#pragma unroll
  for (int r = 0; r < 8; ++r) {
    int pc = mt * 16 + r + (half << 3);
    int po = (pc / 12 + 1) * 14 + (pc % 12 + 1);        // padded pixel index
    float v0 = acc0[r] + bias0; v0 = v0 > 0.0f ? v0 : 0.0f;
    float v1 = acc1[r] + bias1; v1 = v1 > 0.0f ? v1 : 0.0f;
    y1p[(b * YPAD + po) * COUT + co0] = f2bf(v0);
    y1p[(b * YPAD + po) * COUT + co1] = f2bf(v1);
  }
}

// ---------------------------------------------------------------------------
// Expert conv 3x3 pad1 (64->64) + BN + ReLU + routed accumulate into d_out.
// Branch-free: reads padded y1p.  grid 9216, block 64 (2 waves x 2 n-tiles).
// ---------------------------------------------------------------------------
__global__ void k_conv2(const __bf16* __restrict__ y1p,
                        const __bf16* __restrict__ Bc,
                        const float* __restrict__ bc,
                        const float* __restrict__ bng,
                        const float* __restrict__ bnb,
                        const float* __restrict__ bnm,
                        const float* __restrict__ bnv,
                        const float* __restrict__ wgt,
                        float* __restrict__ out, int e) {
  const int b  = blockIdx.x / 9;
  const int mt = blockIdx.x % 9;
  const float wq = wgt[b * 8 + e];
  if (wq == 0.0f) return;                 // block-uniform early out

  const int lane = threadIdx.x & 31;
  const int wv   = threadIdx.x >> 5;
  const int nb0  = wv * 2, nb1 = nb0 + 1;
  const int half = lane >> 4;
  const int pix  = mt * 16 + (lane & 15);
  const int oh   = pix / 12, ow = pix % 12;

  v8f acc0 = {}, acc1 = {};
#pragma unroll
  for (int kh = 0; kh < 3; ++kh) {
#pragma unroll
    for (int kw = 0; kw < 3; ++kw) {
      // padded coords: (oh+kh-1)+1 = oh+kh in [0,13]
      const __bf16* rowBase =
          y1p + (b * YPAD + (oh + kh) * 14 + (ow + kw)) * COUT;
      const int tap = kh * 3 + kw;
#pragma unroll
      for (int kb = 0; kb < 2; ++kb) {
        v16bf a  = load_a_frag(rowBase + kb * 32, half);
        v16bf b0 = load_b_frag(Bc + (((e * 9 + tap) * 2 + kb) * 4 + nb0) * 512, lane);
        v16bf b1 = load_b_frag(Bc + (((e * 9 + tap) * 2 + kb) * 4 + nb1) * 512, lane);
        acc0 = wmma_bf16(a, b0, acc0);
        acc1 = wmma_bf16(a, b1, acc1);
      }
    }
  }
#pragma unroll
  for (int j = 0; j < 2; ++j) {
    const int co = (j ? nb1 : nb0) * 16 + (lane & 15);
    const int ec = e * COUT + co;
    const float bias  = bc[ec];
    const float scale = bng[ec] * rsqrtf(bnv[ec] + 1e-5f);
    const float mean  = bnm[ec];
    const float beta  = bnb[ec];
    const v8f& acc = j ? acc1 : acc0;
#pragma unroll
    for (int r = 0; r < 8; ++r) {
      int pc = mt * 16 + r + (half << 3);
      float v = acc[r] + bias;
      v = (v - mean) * scale + beta;
      v = v > 0.0f ? v : 0.0f;
      float* o = out + ((b * COUT + co) * OPIX + pc);
      *o += wq * v;  // tiles partition (b,co,pix); expert launches serialized
    }
  }
}

// ---------------------------------------------------------------------------
// Host-side launch
// ---------------------------------------------------------------------------
extern "C" void kernel_launch(void* const* d_in, const int* in_sizes, int n_in,
                              void* d_out, int out_size, void* d_ws, size_t ws_size,
                              hipStream_t stream) {
  const float* x    = (const float*)d_in[0];
  const float* gw   = (const float*)d_in[1];
  const float* gb   = (const float*)d_in[2];
  const float* fc1w = (const float*)d_in[3];
  const float* fc1b = (const float*)d_in[4];
  const float* fc2w = (const float*)d_in[5];
  const float* fc2b = (const float*)d_in[6];
  const float* wd   = (const float*)d_in[7];
  const float* bd   = (const float*)d_in[8];
  const float* wc   = (const float*)d_in[9];
  const float* bc   = (const float*)d_in[10];
  const float* bng  = (const float*)d_in[11];
  const float* bnb  = (const float*)d_in[12];
  const float* bnm  = (const float*)d_in[13];
  const float* bnv  = (const float*)d_in[14];
  float* out = (float*)d_out;

  char* ws = (char*)d_ws;
  __bf16* xp  = (__bf16*)(ws + 0);          // 16.8 MB  padded x NHWC bf16 [1024][8][8][128]
  __bf16* hp  = (__bf16*)(ws + 16777216);   //  9.4 MB  gate conv out [1024][36][128]
  __bf16* y1p = (__bf16*)(ws + 26214400);   // 25.7 MB  padded deconv out [1024][14][14][64]
  __bf16* pfl = (__bf16*)(ws + 51904512);   //  2.4 MB  pooled/flattened
  __bf16* zp  = (__bf16*)(ws + 54263808);   //  0.5 MB  fc1 out
  float*  wgt = (float*) (ws + 54788096);   //  32 KB   dense routing weights
  __bf16* Bg  = (__bf16*)(ws + 54820864);   //  gate conv weight fragments
  __bf16* Bf1 = (__bf16*)(ws + 55115776);   //  fc1 weight fragments
  __bf16* Bd  = (__bf16*)(ws + 55705600);   //  deconv weight fragments
  __bf16* Bc  = (__bf16*)(ws + 56885248);   //  conv2 weight fragments
  (void)in_sizes; (void)n_in; (void)out_size; (void)ws_size;

  // 0. zero output (routed accumulate target) and y1p (zero halos, det. state)
  k_zero_f32<<<4096, 256, 0, stream>>>(out, B_ * COUT * OPIX);
  k_zero_u32<<<4096, 256, 0, stream>>>((unsigned*)y1p, B_ * YPAD * COUT / 2);

  // 1. packs
  k_pack_x  <<<(B_ * XPAD * CIN + 255) / 256, 256, 0, stream>>>(x, xp);
  k_pack_gw <<<576,  256, 0, stream>>>(gw, Bg);
  k_pack_fc1<<<1152, 256, 0, stream>>>(fc1w, Bf1);
  k_pack_wd <<<2304, 256, 0, stream>>>(wd, Bd);
  k_pack_wc <<<1152, 256, 0, stream>>>(wc, Bc);

  // 2. gate
  k_gate_conv<<<2304, 128, 0, stream>>>(xp, Bg, gb, hp);
  k_maxpool  <<<(B_ * 1152 + 255) / 256, 256, 0, stream>>>(hp, pfl);
  k_fc1      <<<dim3(64, 2), 128, 0, stream>>>(pfl, Bf1, fc1b, zp);
  k_route    <<<4, 256, 0, stream>>>(zp, fc2w, fc2b, wgt);

  // 3. experts (top-2 routed; unrouted batches skip at block granularity)
  for (int e = 0; e < E_; ++e) {
    k_deconv<<<B_ * 9, 64, 0, stream>>>(xp, Bd, bd, wgt, y1p, e);
    k_conv2 <<<B_ * 9, 64, 0, stream>>>(y1p, Bc, bc, bng, bnb, bnm, bnv,
                                        wgt, out, e);
  }
}
